// CrossAttention2d_5669356834677
// MI455X (gfx1250) — matrix operand order repaired and checked by reading, not compiled, not run on hardware
//
#include <hip/hip_runtime.h>
#include <hip/hip_bf16.h>
#include <stdint.h>

// ---------------------------------------------------------------------------
// CDNA5 (gfx1250, wave32) cross-attention pipeline:
//   v_wmma_f32_16x16x32_f16 matrix cores + global_load_async_to_lds_b128
//   (ASYNCcnt) double-buffered operand staging.
// ---------------------------------------------------------------------------

typedef __attribute__((ext_vector_type(16))) _Float16 v16h;
typedef __attribute__((ext_vector_type(8)))  _Float16 v8h;
typedef __attribute__((ext_vector_type(8)))  float    v8f;

#define NB     16
#define C_DEC  1024
#define HW     1024      // 32*32
#define C_ENC  768
#define S_LEN  256
#define NHEAD  16
#define DHEAD  64

// ---- WMMA / async helpers -------------------------------------------------

__device__ __forceinline__ v8f wmma32(v16h a, v16h b, v8f c) {
  // 8 args: (neg_a, A, neg_b, B, c_mod, C, reuse_a, reuse_b)
  return __builtin_amdgcn_wmma_f32_16x16x32_f16(false, a, false, b, (short)0, c,
                                                false, false);
}

// Async DMA: global -> LDS, 16 bytes per lane, tracked by ASYNCcnt.
__device__ __forceinline__ void async_load_b128(unsigned lds_addr,
                                                const void* gptr) {
  asm volatile("global_load_async_to_lds_b128 %0, %1, off"
               :: "v"(lds_addr), "v"((unsigned long long)(uintptr_t)gptr)
               : "memory");
}
__device__ __forceinline__ unsigned lds_off(const void* p) {
  return (unsigned)(uintptr_t)p;   // generic-pointer low bits = LDS offset
}

// A fragment order used by this code (not AMD's table order): lane%16 = M row; lanes 0-15 hold K=0..15
// (VGPR v -> pair {2v,2v+1}), lanes 16-31 hold K=16..31 (matches B).
__device__ __forceinline__ int kpairA(int v, int half) {
  return 2 * v + 16 * half;
}
// B fragment order used by this code (not AMD's table order): lane%16 = N column; lanes 0-15 hold K=0..15
// (VGPR v -> pair {2v,2v+1}), lanes 16-31 hold K=16..31.
__device__ __forceinline__ int kpairB(int v, int half) {
  return 2 * v + 16 * half;
}

__device__ __forceinline__ v16h load_a_frag(const _Float16* row_base,
                                            int rstride, int kofs) {
  int lane = threadIdx.x & 31;
  int m = lane & 15, half = lane >> 4;
  union { unsigned u[8]; v16h h; } f;
  const _Float16* p = row_base + (size_t)m * rstride + kofs;
#pragma unroll
  for (int v = 0; v < 8; ++v)
    f.u[v] = *(const unsigned*)(p + kpairA(v, half));
  return f.h;
}

__device__ __forceinline__ v16h load_b_frag(const _Float16* col_base,
                                            int cstride, int kofs) {
  int lane = threadIdx.x & 31;
  int nn = lane & 15, half = lane >> 4;
  union { unsigned u[8]; v16h h; } f;
  const _Float16* p = col_base + (size_t)nn * cstride + kofs;
#pragma unroll
  for (int v = 0; v < 8; ++v)
    f.u[v] = *(const unsigned*)(p + kpairB(v, half));
  return f.h;
}

// ---- weight fp32 -> f16 ---------------------------------------------------

__global__ __launch_bounds__(256) void cvt_f32_f16_kernel(
    const float* __restrict__ in, _Float16* __restrict__ out, int nelem) {
  int i = blockIdx.x * 256 + threadIdx.x;
  if (i < nelem) out[i] = (_Float16)in[i];
}

// ---- LayerNorm over channels of x; output transposed (n, hw, c) f16 -------

__global__ __launch_bounds__(256) void ln_x_kernel(
    const float* __restrict__ x, const float* __restrict__ gamma,
    const float* __restrict__ beta, _Float16* __restrict__ ndT) {
  int n = blockIdx.y;
  int p = blockIdx.x * 256 + threadIdx.x;
  const float* xb = x + (size_t)n * C_DEC * HW + p;
  float sum = 0.f, sq = 0.f;
  for (int c = 0; c < C_DEC; ++c) {
    float v = xb[(size_t)c * HW];
    sum += v; sq += v * v;
  }
  float mu   = sum * (1.0f / C_DEC);
  float var  = sq * (1.0f / C_DEC) - mu * mu;
  float rstd = __frsqrt_rn(var + 1e-5f);
  _Float16* ob = ndT + ((size_t)n * HW + p) * C_DEC;
  for (int c = 0; c < C_DEC; ++c) {
    float v = xb[(size_t)c * HW];   // second pass hits L2 (x = 64MB < 192MB)
    ob[c] = (_Float16)(((v - mu) * rstd) * gamma[c] + beta[c]);
  }
}

// ---- LayerNorm over last dim of enc; output (n*s, 768) f16 -----------------

__global__ __launch_bounds__(256) void ln_enc_kernel(
    const float* __restrict__ enc, const float* __restrict__ gamma,
    const float* __restrict__ beta, _Float16* __restrict__ out) {
  int row  = blockIdx.x * 8 + (threadIdx.x >> 5);
  int lane = threadIdx.x & 31;
  const float* eb = enc + (size_t)row * C_ENC;
  float sum = 0.f, sq = 0.f;
  for (int c = lane; c < C_ENC; c += 32) { float v = eb[c]; sum += v; sq += v * v; }
#pragma unroll
  for (int off = 16; off >= 1; off >>= 1) {
    sum += __shfl_xor(sum, off, 32);
    sq  += __shfl_xor(sq,  off, 32);
  }
  float mu   = sum * (1.0f / C_ENC);
  float var  = sq * (1.0f / C_ENC) - mu * mu;
  float rstd = __frsqrt_rn(var + 1e-5f);
  _Float16* ob = out + (size_t)row * C_ENC;
  for (int c = lane; c < C_ENC; c += 32)
    ob[c] = (_Float16)(((eb[c] - mu) * rstd) * gamma[c] + beta[c]);
}

// ---- Generic batched WMMA GEMM: C[b] = A[b] (MxK, f16) * W^T (W is NxK) ----
// Block tile 64(M) x 128(N), K-step 32, 8 waves. Operand tiles staged with
// global_load_async_to_lds_b128 and double-buffered; s_wait_asynccnt overlaps
// the next tile's HBM fetch with the current 4 WMMAs.
// RES_OUT=false: C f16 row-major. RES_OUT=true: fp32 (b, N, M) + residual.

template <bool RES_OUT>
__global__ __launch_bounds__(256) void gemm_wmma_kernel(
    const _Float16* __restrict__ A, size_t strideA, int lda,
    const _Float16* __restrict__ W, int ldw, const float* __restrict__ bias,
    const float* __restrict__ xres, _Float16* __restrict__ Ch, size_t strideC,
    float* __restrict__ Cf, int M, int N, int K) {
  __shared__ _Float16 Ash[2][64 * 32];
  __shared__ _Float16 Bsh[2][128 * 32];
  int b = blockIdx.z;
  int m0 = blockIdx.y * 64;
  int n0 = blockIdx.x * 128;
  int tid = threadIdx.x;
  int w = tid >> 5;
  int mrow = (w & 3) * 16;     // wave M offset within tile
  int ng   = (w >> 2) * 64;    // wave N group offset (0 or 64)
  const _Float16* Ab = A + (size_t)b * strideA;

  // staging coordinates (3 async b128 per thread per K-step)
  int ar = tid >> 2,           ac  = (tid & 3) * 8;          // 256 A chunks
  int br0 = tid >> 2,          bc0 = (tid & 3) * 8;          // B chunks 0..255
  int br1 = (tid + 256) >> 2,  bc1 = (tid & 3) * 8;          // B chunks 256..511

  v8f acc[4] = {};
  int steps = K / 32;

  auto stage = [&](int kc) {
    int k0 = kc * 32, buf = kc & 1;
    async_load_b128(lds_off(&Ash[buf][ar * 32 + ac]),
                    &Ab[(size_t)(m0 + ar) * lda + k0 + ac]);
    async_load_b128(lds_off(&Bsh[buf][br0 * 32 + bc0]),
                    &W[(size_t)(n0 + br0) * ldw + k0 + bc0]);
    async_load_b128(lds_off(&Bsh[buf][br1 * 32 + bc1]),
                    &W[(size_t)(n0 + br1) * ldw + k0 + bc1]);
  };

  stage(0);
  for (int kc = 0; kc < steps; ++kc) {
    int buf = kc & 1;
    if (kc + 1 < steps) {
      stage(kc + 1);
      // async loads complete in order: <=3 pending == stage(kc) done
      asm volatile("s_wait_asynccnt 0x3" ::: "memory");
    } else {
      asm volatile("s_wait_asynccnt 0x0" ::: "memory");
    }
    if (kc + 2 < steps) {  // prefetch K-step+2 into L2 (global_prefetch_b8)
      __builtin_prefetch(&Ab[(size_t)(m0 + ar) * lda + (kc + 2) * 32 + ac], 0, 0);
      __builtin_prefetch(&W[(size_t)(n0 + br0) * ldw + (kc + 2) * 32 + bc0], 0, 0);
    }
    __syncthreads();

    v16h afrag = load_a_frag(&Ash[buf][mrow * 32], 32, 0);
#pragma unroll
    for (int nt = 0; nt < 4; ++nt) {
      v16h bfrag = load_b_frag(&Bsh[buf][(ng + nt * 16) * 32], 32, 0);
      acc[nt] = wmma32(afrag, bfrag, acc[nt]);
    }
    __syncthreads();   // buf may be overwritten by stage(kc+2) next iteration
  }

  int lane = tid & 31;
  int col16 = lane & 15, half = lane >> 4;
#pragma unroll
  for (int nt = 0; nt < 4; ++nt) {
    int col = n0 + ng + nt * 16 + col16;
    float bv = bias[col];
#pragma unroll
    for (int r = 0; r < 8; ++r) {
      int row = m0 + mrow + r + 8 * half;
      float vout = acc[nt][r] + bv;
      if constexpr (RES_OUT) {
        size_t idx = ((size_t)b * N + col) * M + row;   // (n, c, hw)
        Cf[idx] = xres[idx] + vout;
      } else {
        Ch[(size_t)b * strideC + (size_t)row * N + col] = (_Float16)vout;
      }
    }
  }
}

// ---- Fused attention per (n, head, 64 q-rows) ------------------------------
// Q: (n, hw, 1024) f16; KV: (n, s, 2048) f16 (K cols 0..1023, V cols 1024..).
// K staged via async DMA; V transposed into LDS with 32-bit pair stores.
// Each of 4 waves: 16x256 logit strip (32 WMMAs), register softmax with
// 16-lane shuffle reductions, P bounced through LDS (reusing K's space),
// then 32 WMMAs for P*V.

__global__ __launch_bounds__(128) void attn_kernel(
    const _Float16* __restrict__ Q, const _Float16* __restrict__ KV,
    const int* __restrict__ padding, _Float16* __restrict__ Y) {
  __shared__ _Float16 Ksh[S_LEN * DHEAD];   // K[s][d]  (32 KB) -> later P slots
  __shared__ _Float16 VshT[DHEAD * S_LEN];  // V^T[d][s] (32 KB)

  int n = blockIdx.z, h = blockIdx.y, qt = blockIdx.x;
  int tid = threadIdx.x, w = tid >> 5, lane = tid & 31;
  int col16 = lane & 15, half = lane >> 4;

  const _Float16* Kg = KV + (size_t)n * S_LEN * 2048 + h * DHEAD;
  const _Float16* Vg = Kg + 1024;

  // K: async DMA straight into LDS (2048 x b128 chunks, 16 per thread)
#pragma unroll
  for (int i = 0; i < 16; ++i) {
    int chunk = tid + i * 128;
    int s = chunk >> 3, d8 = (chunk & 7) * 8;
    async_load_b128(lds_off(&Ksh[s * DHEAD + d8]), &Kg[(size_t)s * 2048 + d8]);
  }
  // V: transpose via VGPRs; interleave row pairs so stores are 32-bit
#pragma unroll
  for (int i = 0; i < 8; ++i) {
    int task = tid + i * 128;          // 1024 tasks: 128 s-pairs x 8 d-chunks
    int s = (task >> 3) * 2, d8 = (task & 7) * 8;
    v8h v0 = *(const v8h*)&Vg[(size_t)s * 2048 + d8];
    v8h v1 = *(const v8h*)&Vg[(size_t)(s + 1) * 2048 + d8];
#pragma unroll
    for (int e = 0; e < 8; ++e) {
      union { _Float16 h[2]; unsigned u; } pr;
      pr.h[0] = v0[e]; pr.h[1] = v1[e];
      *(unsigned*)&VshT[(d8 + e) * S_LEN + s] = pr.u;
    }
  }
  asm volatile("s_wait_asynccnt 0x0" ::: "memory");
  __syncthreads();

  // Q fragments for this wave's 16 rows (K = 64 -> two chained WMMAs)
  int p0 = qt * 64 + w * 16;
  const _Float16* Qg = Q + ((size_t)n * HW + p0) * C_DEC + h * DHEAD;
  v16h qa0 = load_a_frag(Qg, C_DEC, 0);
  v16h qa1 = load_a_frag(Qg, C_DEC, 32);

  // logits: 16 N-tiles of 16 kv columns
  v8f accS[16];
#pragma unroll
  for (int t = 0; t < 16; ++t) {
    v8f a = {};
    v16h b0 = load_b_frag(&Ksh[(t * 16) * DHEAD], DHEAD, 0);
    v16h b1 = load_b_frag(&Ksh[(t * 16) * DHEAD], DHEAD, 32);
    a = wmma32(qa0, b0, a);
    a = wmma32(qa1, b1, a);
    accS[t] = a;
  }

  // scale (1/sqrt(64)) and padding mask
#pragma unroll
  for (int t = 0; t < 16; ++t) {
    float pen = 10000.0f * (float)padding[n * S_LEN + t * 16 + col16];
#pragma unroll
    for (int r = 0; r < 8; ++r) accS[t][r] = accS[t][r] * 0.125f - pen;
  }

  // softmax: each row lives in a 16-lane group (fixed half), across 16 tiles
#pragma unroll
  for (int r = 0; r < 8; ++r) {
    float m = -3.0e38f;
#pragma unroll
    for (int t = 0; t < 16; ++t) m = fmaxf(m, accS[t][r]);
#pragma unroll
    for (int off = 8; off >= 1; off >>= 1) m = fmaxf(m, __shfl_xor(m, off, 32));
    float sum = 0.0f;
#pragma unroll
    for (int t = 0; t < 16; ++t) {
      float e = __expf(accS[t][r] - m);
      accS[t][r] = e;
      sum += e;
    }
#pragma unroll
    for (int off = 8; off >= 1; off >>= 1) sum += __shfl_xor(sum, off, 32);
    float inv = __frcp_rn(sum);
#pragma unroll
    for (int t = 0; t < 16; ++t) accS[t][r] *= inv;
  }

  __syncthreads();                       // everyone done reading Ksh
  _Float16* Psh = Ksh + w * 16 * S_LEN;  // this wave's 16x256 P slot
#pragma unroll
  for (int t = 0; t < 16; ++t)
#pragma unroll
    for (int r = 0; r < 8; ++r)
      Psh[(r + 8 * half) * S_LEN + t * 16 + col16] = (_Float16)accS[t][r];
  __syncthreads();

  // Y strip = P (16x256) * V (256x64): 4 N-tiles x 8 K-chunks
  _Float16* Yb = Y + ((size_t)n * HW + p0) * C_DEC + h * DHEAD;
#pragma unroll
  for (int nt = 0; nt < 4; ++nt) {
    v8f acc = {};
#pragma unroll
    for (int kc = 0; kc < 8; ++kc) {
      v16h ap = load_a_frag(Psh, S_LEN, kc * 32);
      v16h bv = load_b_frag(&VshT[(nt * 16) * S_LEN], S_LEN, kc * 32);
      acc = wmma32(ap, bv, acc);
    }
#pragma unroll
    for (int r = 0; r < 8; ++r) {
      int row = r + 8 * half;
      Yb[(size_t)row * C_DEC + nt * 16 + col16] = (_Float16)acc[r];
    }
  }
}

// ---------------------------------------------------------------------------

extern "C" void kernel_launch(void* const* d_in, const int* in_sizes, int n_in,
                              void* d_out, int out_size, void* d_ws,
                              size_t ws_size, hipStream_t stream) {
  const float* x         = (const float*)d_in[0];
  const float* enc       = (const float*)d_in[1];
  const int*   padding   = (const int*)d_in[2];
  const float* gamma_dec = (const float*)d_in[3];
  const float* beta_dec  = (const float*)d_in[4];
  const float* gamma_enc = (const float*)d_in[5];
  const float* beta_enc  = (const float*)d_in[6];
  const float* Wq        = (const float*)d_in[7];
  const float* bq        = (const float*)d_in[8];
  const float* Wkv       = (const float*)d_in[9];
  const float* bkv       = (const float*)d_in[10];
  const float* Wo        = (const float*)d_in[11];
  const float* bo        = (const float*)d_in[12];
  float* out = (float*)d_out;

  char* ws = (char*)d_ws;
  size_t off = 0;
  auto alloc = [&](size_t bytes) -> void* {
    void* p = ws + off;
    off += (bytes + 255) & ~(size_t)255;
    return p;
  };
  _Float16* ndT  = (_Float16*)alloc((size_t)NB * HW * C_DEC * 2);
  _Float16* qb   = (_Float16*)alloc((size_t)NB * HW * C_DEC * 2);
  _Float16* encn = (_Float16*)alloc((size_t)NB * S_LEN * C_ENC * 2);
  _Float16* kvb  = (_Float16*)alloc((size_t)NB * S_LEN * 2 * C_DEC * 2);
  _Float16* yb   = (_Float16*)alloc((size_t)NB * HW * C_DEC * 2);
  _Float16* WqH  = (_Float16*)alloc((size_t)C_DEC * C_DEC * 2);
  _Float16* WkvH = (_Float16*)alloc((size_t)2 * C_DEC * C_ENC * 2);
  _Float16* WoH  = (_Float16*)alloc((size_t)C_DEC * C_DEC * 2);
  (void)ws_size; (void)in_sizes; (void)n_in; (void)out_size;

  // weights fp32 -> f16
  cvt_f32_f16_kernel<<<(C_DEC * C_DEC + 255) / 256, 256, 0, stream>>>(
      Wq, WqH, C_DEC * C_DEC);
  cvt_f32_f16_kernel<<<(2 * C_DEC * C_ENC + 255) / 256, 256, 0, stream>>>(
      Wkv, WkvH, 2 * C_DEC * C_ENC);
  cvt_f32_f16_kernel<<<(C_DEC * C_DEC + 255) / 256, 256, 0, stream>>>(
      Wo, WoH, C_DEC * C_DEC);

  // layernorms
  ln_x_kernel<<<dim3(HW / 256, NB), 256, 0, stream>>>(x, gamma_dec, beta_dec,
                                                      ndT);
  ln_enc_kernel<<<(NB * S_LEN) / 8, 256, 0, stream>>>(enc, gamma_enc, beta_enc,
                                                      encn);

  // Q = ndT * Wq^T + bq : (b=16) M=1024, N=1024, K=1024
  gemm_wmma_kernel<false><<<dim3(C_DEC / 128, HW / 64, NB), 256, 0, stream>>>(
      ndT, (size_t)HW * C_DEC, C_DEC, WqH, C_DEC, bq, nullptr, qb,
      (size_t)HW * C_DEC, nullptr, HW, C_DEC, C_DEC);

  // KV = encn * Wkv^T + bkv : M=256, N=2048, K=768
  gemm_wmma_kernel<false><<<dim3(2048 / 128, S_LEN / 64, NB), 256, 0, stream>>>(
      encn, (size_t)S_LEN * C_ENC, C_ENC, WkvH, C_ENC, bkv, nullptr, kvb,
      (size_t)S_LEN * 2048, nullptr, S_LEN, 2048, C_ENC);

  // attention: grid (q-tiles, heads, batch)
  attn_kernel<<<dim3(HW / 64, NHEAD, NB), 128, 0, stream>>>(qb, kvb, padding,
                                                            yb);

  // out = x + yb * Wo^T + bo  (fp32, NCHW layout)
  gemm_wmma_kernel<true><<<dim3(C_DEC / 128, HW / 64, NB), 256, 0, stream>>>(
      yb, (size_t)HW * C_DEC, C_DEC, WoH, C_DEC, bo, x, nullptr, 0, out, HW,
      C_DEC, C_DEC);
}
